// GraphAttentionLayer_72301479461285
// MI455X (gfx1250) — compile-verified, matrix operand order checked by reading, not run on hardware
//
#include <hip/hip_runtime.h>
#include <stdint.h>

// ---------------- CDNA5 vector types ----------------
typedef __attribute__((ext_vector_type(16))) __bf16        v16bf;
typedef __attribute__((ext_vector_type(8)))  float         v8f;
typedef __attribute__((ext_vector_type(8)))  unsigned int  v8u;

#define N_NODES 4096
#define FDIM    64
#define EDIM    8
#define ALPHA   0.2f
#define NEGINF  (-9.0e15f)

static __device__ __forceinline__ unsigned int f32_to_bf16(float f) {
    unsigned int u = __builtin_bit_cast(unsigned int, f);
    u += 0x7FFFu + ((u >> 16) & 1u);   // round-to-nearest-even (data is non-NaN)
    return u >> 16;
}

// ---------------------------------------------------------------------------
// Prep: h = x @ W (per-row), s1 = h.a1, s2 = h.a2, hT = bf16 transpose of h.
// ---------------------------------------------------------------------------
__global__ __launch_bounds__(256)
void gat_prep(const float* __restrict__ x, const float* __restrict__ W,
              const float* __restrict__ a, float* __restrict__ s1,
              float* __restrict__ s2, unsigned short* __restrict__ hT) {
    __shared__ float Wl[FDIM * FDIM];
    const int tid = threadIdx.x;
    for (int i = tid; i < FDIM * FDIM; i += 256) Wl[i] = W[i];
    __syncthreads();

    const int row = blockIdx.x * 256 + tid;
    float xr[FDIM];
    const float4* xp = (const float4*)(x + (size_t)row * FDIM);
#pragma unroll
    for (int i = 0; i < FDIM / 4; ++i) {
        float4 v = xp[i];
        xr[4 * i + 0] = v.x; xr[4 * i + 1] = v.y;
        xr[4 * i + 2] = v.z; xr[4 * i + 3] = v.w;
    }
    float s1a = 0.f, s2a = 0.f;
    for (int c = 0; c < FDIM; ++c) {
        float acc = 0.f;
#pragma unroll
        for (int k = 0; k < FDIM; ++k) acc = fmaf(xr[k], Wl[k * FDIM + c], acc);
        s1a = fmaf(acc, a[c], s1a);
        s2a = fmaf(acc, a[FDIM + c], s2a);
        hT[(size_t)c * N_NODES + row] = (unsigned short)f32_to_bf16(acc);
    }
    s1[row] = s1a;
    s2[row] = s2a;
}

// ---------------------------------------------------------------------------
// Main fused kernel: scores + mask + online softmax + (P @ h) via WMMA + ELU.
// 1 block = 16 output rows; 8 waves split the 4096 columns into 32-col chunks.
// Lane = column while streaming edge_feats (coalesced); scores bounce through
// LDS once so the softmax/exp/pack all happen in the WMMA A-operand layout.
// ---------------------------------------------------------------------------
__global__ __launch_bounds__(256)
void gat_main(const float* __restrict__ ef, const int* __restrict__ adj,
              const float* __restrict__ a, const float* __restrict__ s1,
              const float* __restrict__ s2, const unsigned short* __restrict__ hT,
              float* __restrict__ out) {
    __shared__ __align__(16) float sbuf[8][16 * 32];   // raw scores, 16 KB
    __shared__ __align__(16) float scalebuf[8][16];    // per-row rescale factors
    __shared__ float accbuf[8][16][FDIM];              // 32 KB
    __shared__ float mbuf[8][16];
    __shared__ float lbuf[8][16];

    const int tid  = threadIdx.x;
    const int wave = tid >> 5;
    const int lane = tid & 31;
    const int half = lane >> 4;     // 0: lanes 0-15, 1: lanes 16-31
    const int l15  = lane & 15;     // this lane owns softmax state of row l15
    const int r0   = blockIdx.x * 16;

    float a3[EDIM];
#pragma unroll
    for (int i = 0; i < EDIM; ++i) a3[i] = a[2 * FDIM + i];    // uniform -> SGPRs
    float s1v[16];
#pragma unroll
    for (int i = 0; i < 16; ++i) s1v[i] = s1[r0 + i];          // uniform -> SGPRs

    float m_run = -3.0e38f;   // running max of row l15 (synced across lane pair)
    float l_run = 0.f;        // running sum-exp of row l15
    v8f acc[4];
#pragma unroll
    for (int t = 0; t < 4; ++t)
#pragma unroll
        for (int v = 0; v < 8; ++v) acc[t][v] = 0.f;

    for (int ci = 0; ci < 16; ++ci) {                 // 16 chunks * 8 waves * 32 = 4096
        const int colbase = (ci * 8 + wave) * 32;
        const int col     = colbase + lane;           // lane = column -> coalesced
        const float s2v   = s2[col];

        // ---- phase 1: stream edge_feats/adj, raw masked scores into LDS ----
#pragma unroll
        for (int j = 0; j < 16; ++j) {
            const int row = r0 + j;
            const float4* efp = (const float4*)(ef + ((size_t)row * N_NODES + col) * EDIM);
            float4 e0 = efp[0], e1 = efp[1];          // 32B/lane, 32B stride: coalesced
            float s3 = e0.x * a3[0] + e0.y * a3[1] + e0.z * a3[2] + e0.w * a3[3]
                     + e1.x * a3[4] + e1.y * a3[5] + e1.z * a3[6] + e1.w * a3[7];
            float e = s1v[j] + s2v + s3;
            e = e > 0.f ? e : ALPHA * e;                              // LeakyReLU
            const int ad = adj[(size_t)row * N_NODES + col];          // coalesced b32
            sbuf[wave][j * 32 + lane] = ad > 0 ? e : NEGINF;
        }
        asm volatile("s_wait_dscnt 0" ::: "memory");

        // ---- phase 2: read back this lane's A-layout slice of the 16x32 tile ----
        // lanes 0-15 : row=lane,   K{0..7,16..23}; lanes 16-31: row=lane-16, K{8..15,24..31}
        const int k0 = half * 8;
        float scv[16];
        {
            const float4* spA = (const float4*)&sbuf[wave][l15 * 32 + k0];
            const float4* spB = (const float4*)&sbuf[wave][l15 * 32 + 16 + k0];
            float4 q0 = spA[0], q1 = spA[1], q2 = spB[0], q3 = spB[1];
            scv[0]  = q0.x; scv[1]  = q0.y; scv[2]  = q0.z; scv[3]  = q0.w;
            scv[4]  = q1.x; scv[5]  = q1.y; scv[6]  = q1.z; scv[7]  = q1.w;
            scv[8]  = q2.x; scv[9]  = q2.y; scv[10] = q2.z; scv[11] = q2.w;
            scv[12] = q3.x; scv[13] = q3.y; scv[14] = q3.z; scv[15] = q3.w;
        }

        // ---- phase 3: online softmax for row l15 (lane-local + 1 pair-swap) ----
        float mx = scv[0];
#pragma unroll
        for (int i = 1; i < 16; ++i) mx = fmaxf(mx, scv[i]);
        mx = fmaxf(mx, __shfl_xor(mx, 16));           // merge the two K-halves
        const float m_new = fmaxf(m_run, mx);
        const float f = __expf(m_run - m_new);

        float p[16];
        float sm = 0.f;
#pragma unroll
        for (int i = 0; i < 16; ++i) { p[i] = __expf(scv[i] - m_new); sm += p[i]; }
        sm += __shfl_xor(sm, 16);                     // merge the two K-halves
        l_run = l_run * f + sm;
        m_run = m_new;
        if (half == 0) scalebuf[wave][l15] = f;       // broadcast per-row scale
        asm volatile("s_wait_dscnt 0" ::: "memory");

        // ---- phase 4: rescale accumulator (C layout: VGPR v = row v + half*8) ----
        {
            const float4* scp = (const float4*)&scalebuf[wave][half * 8];
            float4 f0 = scp[0], f1 = scp[1];
            float fs[8] = { f0.x, f0.y, f0.z, f0.w, f1.x, f1.y, f1.z, f1.w };
#pragma unroll
            for (int v = 0; v < 8; ++v)
#pragma unroll
                for (int t = 0; t < 4; ++t) acc[t][v] *= fs[v];
        }

        // ---- phase 5: pack P into bf16 A operand (already in lane layout) ----
        v8u au;
#pragma unroll
        for (int i = 0; i < 8; ++i)
            au[i] = f32_to_bf16(p[2 * i]) | (f32_to_bf16(p[2 * i + 1]) << 16);
        const v16bf A = __builtin_bit_cast(v16bf, au);

        // ---- phase 6: P @ h via WMMA (B from bf16 hT, K-contiguous, L2-resident) ----
#pragma unroll
        for (int nt = 0; nt < 4; ++nt) {
            const int coln = nt * 16 + l15;
            const unsigned short* hp = hT + (size_t)coln * N_NODES + colbase;
            const uint4* pb0 = (const uint4*)(hp + half * 8);
            const uint4* pb1 = (const uint4*)(hp + 16 + half * 8);
            uint4 blo = *pb0, bhi = *pb1;
            v8u bu;
            bu[0] = blo.x; bu[1] = blo.y; bu[2] = blo.z; bu[3] = blo.w;
            bu[4] = bhi.x; bu[5] = bhi.y; bu[6] = bhi.z; bu[7] = bhi.w;
            const v16bf B = __builtin_bit_cast(v16bf, bu);
            acc[nt] = __builtin_amdgcn_wmma_f32_16x16x32_bf16(
                false, A, false, B, (short)0, acc[nt], false, false);
        }
    }

    // ---- cross-wave softmax merge ----
#pragma unroll
    for (int nt = 0; nt < 4; ++nt)
#pragma unroll
        for (int v = 0; v < 8; ++v)
            accbuf[wave][v + half * 8][nt * 16 + l15] = acc[nt][v];
    if (half == 0) { mbuf[wave][l15] = m_run; lbuf[wave][l15] = l_run; }
    __syncthreads();

    for (int idx = tid; idx < 16 * FDIM; idx += 256) {
        const int r = idx >> 6, c = idx & 63;
        float M = mbuf[0][r];
#pragma unroll
        for (int w = 1; w < 8; ++w) M = fmaxf(M, mbuf[w][r]);
        float den = 0.f, num = 0.f;
#pragma unroll
        for (int w = 0; w < 8; ++w) {
            const float fw = __expf(mbuf[w][r] - M);
            den += lbuf[w][r] * fw;
            num += accbuf[w][r][c] * fw;
        }
        const float v = num / den;
        out[(size_t)(r0 + r) * FDIM + c] = v > 0.f ? v : __expf(v) - 1.f;  // ELU
    }
}

// ---------------------------------------------------------------------------
extern "C" void kernel_launch(void* const* d_in, const int* in_sizes, int n_in,
                              void* d_out, int out_size, void* d_ws, size_t ws_size,
                              hipStream_t stream) {
    const float* x   = (const float*)d_in[0];
    const int*   adj = (const int*)d_in[1];
    const float* ef  = (const float*)d_in[2];
    const float* W   = (const float*)d_in[3];
    const float* a   = (const float*)d_in[4];
    float* out = (float*)d_out;

    char* ws = (char*)d_ws;
    unsigned short* hT = (unsigned short*)ws;                        // 64*4096*2 = 512 KB
    float* s1 = (float*)(ws + (size_t)FDIM * N_NODES * sizeof(unsigned short));
    float* s2 = s1 + N_NODES;                                        // +16 KB each

    gat_prep<<<N_NODES / 256, 256, 0, stream>>>(x, W, a, s1, s2, hT);
    gat_main<<<N_NODES / 16, 256, 0, stream>>>(ef, adj, a, s1, s2, hT, out);
}
